// AGNNLayer_55783035240597
// MI455X (gfx1250) — compile-verified, matrix operand order checked by reading, not run on hardware
//
#include <hip/hip_runtime.h>
#include <hip/hip_fp16.h>

typedef _Float16 v16h __attribute__((ext_vector_type(16)));
typedef _Float16 v8h  __attribute__((ext_vector_type(8)));
typedef _Float16 v4h  __attribute__((ext_vector_type(4)));
typedef float    v8f  __attribute__((ext_vector_type(8)));

#define AGNN_EPS 1e-12f

// ---------------- K0: zero/init accumulators ----------------
__global__ void k_init(float* __restrict__ out, int outN,
                       float* __restrict__ s, unsigned* __restrict__ menc, int N) {
    int i = blockIdx.x * blockDim.x + threadIdx.x;
    if (i < outN) out[i] = 0.0f;
    if (i < N) { s[i] = 0.0f; menc[i] = 0u; }
}

// ---------------- K1: row-normalize x -> f16 ----------------
// one wave (32 lanes) per row of 128 floats
__global__ void k_norm(const float* __restrict__ x, _Float16* __restrict__ xnh, int N) {
    int lane = threadIdx.x & 31;
    int row  = (blockIdx.x * blockDim.x + threadIdx.x) >> 5;
    if (row >= N) return;
    const float4 v = *(const float4*)(x + (size_t)row * 128 + lane * 4);
    float ss = v.x * v.x + v.y * v.y + v.z * v.z + v.w * v.w;
    #pragma unroll
    for (int off = 16; off > 0; off >>= 1) ss += __shfl_xor(ss, off, 32);
    float scale = rsqrtf(fmaxf(ss, AGNN_EPS * AGNN_EPS));
    v4h h = { (_Float16)(v.x * scale), (_Float16)(v.y * scale),
              (_Float16)(v.z * scale), (_Float16)(v.w * scale) };
    *(v4h*)(xnh + (size_t)row * 128 + lane * 4) = h;
}

// ---------------- K2: per-edge cosine logits via WMMA ----------------
// One wave handles 16 edges. A = xn[src] rows (16x128), B = xn[dst] cols (128x16),
// 4 x v_wmma_f32_16x16x32_f16, logits = beta * diag(C).
__global__ void k_logits(const _Float16* __restrict__ xnh,
                         const int* __restrict__ srcIdx,
                         const int* __restrict__ dstIdx,
                         const float* __restrict__ beta,
                         float* __restrict__ logits,
                         int E, long long ET) {
    __shared__ float diag[8][32][8];   // 8 waves/block
    const int wInB = threadIdx.x >> 5;
    const int lane = threadIdx.x & 31;
    const long long wave = (long long)blockIdx.x * 8 + wInB;
    const long long e16  = wave * 16;
    const int m    = lane & 15;
    const int ksel = lane >> 4;        // 0: K 0..7/16..23, 1: K 8..15/24..31 per 32-chunk
    const long long e = e16 + m;

    int sR = 0, dR = 0;
    if (e < (long long)E)      { sR = srcIdx[e]; dR = dstIdx[e]; }
    else if (e < ET)           { sR = dR = (int)(e - (long long)E); }   // self-loop

    const _Float16* abase = xnh + (size_t)sR * 128 + ksel * 8;
    const _Float16* bbase = xnh + (size_t)dR * 128 + ksel * 8;

    v8f c = {};
    #pragma unroll
    for (int kc = 0; kc < 4; ++kc) {
        v8h alo = *(const v8h*)(abase + kc * 32);
        v8h ahi = *(const v8h*)(abase + kc * 32 + 16);
        v8h blo = *(const v8h*)(bbase + kc * 32);
        v8h bhi = *(const v8h*)(bbase + kc * 32 + 16);
        v16h a = __builtin_shufflevector(alo, ahi, 0,1,2,3,4,5,6,7,8,9,10,11,12,13,14,15);
        v16h b = __builtin_shufflevector(blo, bhi, 0,1,2,3,4,5,6,7,8,9,10,11,12,13,14,15);
        c = __builtin_amdgcn_wmma_f32_16x16x32_f16(false, a, false, b, (short)0, c,
                                                   false, false);
    }

    #pragma unroll
    for (int i = 0; i < 8; ++i) diag[wInB][lane][i] = c[i];
    __syncthreads();

    if (lane < 16 && e < ET) {
        // C layout: VGPR v: lanes0-15 -> M=v, N=lane; lanes16-31 -> M=v+8, N=lane-16
        int srcLane = (m < 8) ? m : (m + 16);
        int ci      = (m < 8) ? m : (m - 8);
        logits[e] = beta[0] * diag[wInB][srcLane][ci];
    }
}

// monotone float<->uint encoding for atomicMax-based segment max
__device__ __forceinline__ unsigned encf(float f) {
    unsigned u = __float_as_uint(f);
    return (u & 0x80000000u) ? ~u : (u | 0x80000000u);
}
__device__ __forceinline__ float decf(unsigned u) {
    return (u & 0x80000000u) ? __uint_as_float(u & 0x7FFFFFFFu) : __uint_as_float(~u);
}

// ---------------- K3: segment max over dst ----------------
__global__ void k_segmax(const float* __restrict__ logits, const int* __restrict__ dstIdx,
                         unsigned* __restrict__ menc, int E, long long ET) {
    long long e = (long long)blockIdx.x * blockDim.x + threadIdx.x;
    if (e >= ET) return;
    int d = (e < (long long)E) ? dstIdx[e] : (int)(e - (long long)E);
    atomicMax(menc + d, encf(logits[e]));
}

// ---------------- K4: e = exp(logit - m[dst]); segment sum ----------------
__global__ void k_expsum(float* __restrict__ logits, const int* __restrict__ dstIdx,
                         const unsigned* __restrict__ menc, float* __restrict__ s,
                         int E, long long ET) {
    long long e = (long long)blockIdx.x * blockDim.x + threadIdx.x;
    if (e >= ET) return;
    int d = (e < (long long)E) ? dstIdx[e] : (int)(e - (long long)E);
    float ev = expf(logits[e] - decf(menc[d]));
    logits[e] = ev;                      // overwrite logit with exp value
    unsafeAtomicAdd(s + d, ev);          // native global_atomic_add_f32
}

// ---------------- K5: scatter-aggregate alpha * x[src] -> out[dst] ----------------
// one wave per edge; 4 floats per lane
__global__ void k_agg(const float* __restrict__ x, const float* __restrict__ ev,
                      const float* __restrict__ s,
                      const int* __restrict__ srcIdx, const int* __restrict__ dstIdx,
                      float* __restrict__ out, int E, long long ET) {
    int lane = threadIdx.x & 31;
    long long e = ((long long)blockIdx.x * blockDim.x + threadIdx.x) >> 5;
    if (e >= ET) return;
    int sR, dR;
    if (e < (long long)E) { sR = srcIdx[e]; dR = dstIdx[e]; }
    else                  { sR = dR = (int)(e - (long long)E); }
    float alpha = ev[e] / s[dR];
    const float4 v = *(const float4*)(x + (size_t)sR * 128 + lane * 4);
    float* o = out + (size_t)dR * 128 + lane * 4;
    unsafeAtomicAdd(o + 0, alpha * v.x);
    unsafeAtomicAdd(o + 1, alpha * v.y);
    unsafeAtomicAdd(o + 2, alpha * v.z);
    unsafeAtomicAdd(o + 3, alpha * v.w);
}

// ---------------- K6: ReLU ----------------
__global__ void k_relu(float* __restrict__ out, int n) {
    int i = blockIdx.x * blockDim.x + threadIdx.x;
    if (i < n) out[i] = fmaxf(out[i], 0.0f);
}

extern "C" void kernel_launch(void* const* d_in, const int* in_sizes, int n_in,
                              void* d_out, int out_size, void* d_ws, size_t ws_size,
                              hipStream_t stream) {
    const float* x    = (const float*)d_in[0];
    const float* beta = (const float*)d_in[1];
    const int*   ei   = (const int*)d_in[2];

    const int N = in_sizes[0] / 128;
    const int E = in_sizes[2] / 2;
    const long long ET = (long long)E + (long long)N;

    const int* srcIdx = ei;         // edge_index[0]
    const int* dstIdx = ei + E;     // edge_index[1]
    float* out = (float*)d_out;

    // workspace carve-out (~20 MB total)
    char* ws = (char*)d_ws;
    size_t off = 0;
    _Float16* xnh = (_Float16*)(ws + off);
    off += (size_t)N * 128 * sizeof(_Float16); off = (off + 255) & ~(size_t)255;
    float* logits = (float*)(ws + off);
    off += (size_t)ET * sizeof(float);         off = (off + 255) & ~(size_t)255;
    unsigned* menc = (unsigned*)(ws + off);
    off += (size_t)N * sizeof(unsigned);       off = (off + 255) & ~(size_t)255;
    float* s = (float*)(ws + off);

    const int B = 256;

    // K0: init
    {
        int n = out_size > N ? out_size : N;
        k_init<<<(n + B - 1) / B, B, 0, stream>>>(out, out_size, s, menc, N);
    }
    // K1: normalize (1 wave per row)
    k_norm<<<(N + 7) / 8, B, 0, stream>>>(x, xnh, N);
    // K2: WMMA logits (1 wave per 16 edges, 8 waves per block)
    {
        long long nw = (ET + 15) / 16;
        unsigned blocks = (unsigned)((nw + 7) / 8);
        k_logits<<<blocks, B, 0, stream>>>(xnh, srcIdx, dstIdx, beta, logits, E, ET);
    }
    // K3: segment max
    k_segmax<<<(unsigned)((ET + B - 1) / B), B, 0, stream>>>(logits, dstIdx, menc, E, ET);
    // K4: exp + segment sum
    k_expsum<<<(unsigned)((ET + B - 1) / B), B, 0, stream>>>(logits, dstIdx, menc, s, E, ET);
    // K5: aggregate (1 wave per edge)
    k_agg<<<(unsigned)((ET + 7) / 8), B, 0, stream>>>(x, logits, s, srcIdx, dstIdx, out, E, ET);
    // K6: ReLU
    k_relu<<<(out_size + B - 1) / B, B, 0, stream>>>(out, out_size);
}
